// ExpertsFFN_137438954163
// MI455X (gfx1250) — matrix operand order repaired and checked by reading, not compiled, not run on hardware
//
#include <hip/hip_runtime.h>

// ---------------------------------------------------------------------------
// MoE expert FFN: out = silu(x @ W1^T + b1) @ W2^T + b2, batched over experts.
// E=16, CAP=256, D_MODEL=1024, D_FF=4096, fp32 I/O.
//
// Roofline: ~68.7 GFLOP vs ~545 MB HBM traffic -> ~23us floor at 23.3 TB/s,
// requiring ~3 PFLOPS effective. Only 16-bit WMMA reaches that on MI455X, so
// fp32 is converted to bf16 in flight and the GEMMs run on
// v_wmma_f32_16x16x32_bf16 (f32 accumulate). Intermediate H is bf16 in d_ws.
// Double-buffered LDS: one barrier per K-step, global loads of tile k+1
// overlap the WMMAs of tile k.
// ---------------------------------------------------------------------------

typedef __bf16 bf16;
typedef __attribute__((ext_vector_type(16))) __bf16 v16bf;
typedef __attribute__((ext_vector_type(8)))  __bf16 v8bf;
typedef __attribute__((ext_vector_type(8)))  float  v8f;

#define NEXPERT 16
#define CAPN    256
#define DMODEL  1024
#define DFF     4096

#define BM 128      // block tile M
#define BN 64       // block tile N
#define BK 32       // K step (matches wmma 16x16x32)
#define LDA 40      // padded LDS row stride in bf16 elems (80 B, 16 B aligned)

// Build A fragment (16x32 bf16) for this lane from an LDS tile row.
// ISA 7.12.2: lanes 0-15 (khalf=0) hold K={0..7,16..23}; lanes 16-31 K={8..15,24..31}.
__device__ __forceinline__ v16bf load_a_frag(const bf16* rowp, int khalf) {
    v8bf lo = *(const v8bf*)(rowp + khalf * 8);
    v8bf hi = *(const v8bf*)(rowp + 16 + khalf * 8);
    v16bf a;
#pragma unroll
    for (int i = 0; i < 8; ++i) { a[i] = lo[i]; a[8 + i] = hi[i]; }
    return a;
}

// Build B fragment (32x16 bf16): lane n holds K={khalf*8..+7, 16+khalf*8..+7}, same order as A.
__device__ __forceinline__ v16bf load_b_frag(const bf16* rowp, int khalf) {
    v8bf lo = *(const v8bf*)(rowp + khalf * 8);
    v8bf hi = *(const v8bf*)(rowp + 16 + khalf * 8);
    v16bf b;
#pragma unroll
    for (int i = 0; i < 8; ++i) { b[i] = lo[i]; b[8 + i] = hi[i]; }
    return b;
}

// fast sigmoid: v_exp_f32 + v_rcp_f32, no IEEE divide chain
__device__ __forceinline__ float fast_silu(float v) {
    return v * __builtin_amdgcn_rcpf(1.0f + __expf(-v));
}

// ---------------------------------------------------------------------------
// GEMM1: H[e,b,f] = silu( sum_d X[e,b,d] * W1[e,f,d] + b1[e,f] ), H in bf16.
// M=CAP (256), N=DFF (4096), K=DMODEL (1024).
// ---------------------------------------------------------------------------
__global__ __launch_bounds__(256)
void ffn_gemm1_silu(const float* __restrict__ X,
                    const float* __restrict__ W1,
                    const float* __restrict__ B1,
                    bf16* __restrict__ H) {
    const int e    = blockIdx.z;
    const int nb   = blockIdx.x;          // N tile: DFF/BN
    const int mb   = blockIdx.y;          // M tile: CAP/BM
    const int tid  = threadIdx.x;
    const int lane = tid & 31;
    const int wave = tid >> 5;            // 0..7
    const int waveM = wave >> 1;          // 0..3 -> 32-row strip
    const int waveN = wave & 1;           // 0..1 -> 32-col strip
    const int khalf = lane >> 4;
    const int mrow  = lane & 15;

    __shared__ bf16 As[2][BM][LDA];
    __shared__ bf16 Bs[2][BN][LDA];

    const float* Xe  = X  + (size_t)e * CAPN * DMODEL + (size_t)mb * BM * DMODEL;
    const float* W1e = W1 + (size_t)e * DFF * DMODEL  + (size_t)nb * BN * DMODEL;

    // Stage one BKxBM/BKxBN tile (fp32 -> bf16) into LDS buffer `buf`.
    auto stage = [&](int buf, int k0) {
        // A: 128x32 floats -> 1024 float4; 4 per thread.
#pragma unroll
        for (int it = 0; it < 4; ++it) {
            int idx = tid + it * 256;
            int r   = idx >> 3;                // 8 float4 per row
            int c4  = idx & 7;
            float4 v = ((const float4*)(Xe + (size_t)r * DMODEL + k0))[c4];
            bf16* dst = &As[buf][r][c4 * 4];
            dst[0] = (bf16)v.x; dst[1] = (bf16)v.y;
            dst[2] = (bf16)v.z; dst[3] = (bf16)v.w;
        }
        // B: 64x32 floats -> 512 float4; 2 per thread.
#pragma unroll
        for (int it = 0; it < 2; ++it) {
            int idx = tid + it * 256;
            int r   = idx >> 3;
            int c4  = idx & 7;
            float4 v = ((const float4*)(W1e + (size_t)r * DMODEL + k0))[c4];
            bf16* dst = &Bs[buf][r][c4 * 4];
            dst[0] = (bf16)v.x; dst[1] = (bf16)v.y;
            dst[2] = (bf16)v.z; dst[3] = (bf16)v.w;
        }
    };

    v8f acc[2][2] = {};

    stage(0, 0);
    __syncthreads();

    const int nsteps = DMODEL / BK;
    for (int ks = 0; ks < nsteps; ++ks) {
        const int buf = ks & 1;
        if (ks + 1 < nsteps) stage(buf ^ 1, (ks + 1) * BK);   // overlaps WMMAs

        v16bf afrag[2], bfrag[2];
#pragma unroll
        for (int mi = 0; mi < 2; ++mi)
            afrag[mi] = load_a_frag(&As[buf][waveM * 32 + mi * 16 + mrow][0], khalf);
#pragma unroll
        for (int ni = 0; ni < 2; ++ni)
            bfrag[ni] = load_b_frag(&Bs[buf][waveN * 32 + ni * 16 + mrow][0], khalf);

#pragma unroll
        for (int mi = 0; mi < 2; ++mi)
#pragma unroll
            for (int ni = 0; ni < 2; ++ni)
                acc[mi][ni] = __builtin_amdgcn_wmma_f32_16x16x32_bf16(
                    false, afrag[mi], false, bfrag[ni],
                    (short)0, acc[mi][ni], false, false);
        __syncthreads();
    }

    // Epilogue: bias + SiLU, store bf16 H.
    const float* b1p = B1 + (size_t)e * DFF + (size_t)nb * BN;
    bf16* He = H + (size_t)e * CAPN * DFF + (size_t)mb * BM * DFF + (size_t)nb * BN;
    const int dm = (lane >> 4) * 8;   // C/D layout: m = i + (lane/16)*8
    const int dn = lane & 15;         //             n = lane & 15
#pragma unroll
    for (int mi = 0; mi < 2; ++mi)
#pragma unroll
        for (int ni = 0; ni < 2; ++ni) {
            int nc = waveN * 32 + ni * 16 + dn;
            float bias = b1p[nc];
#pragma unroll
            for (int i = 0; i < 8; ++i) {
                int mr = waveM * 32 + mi * 16 + dm + i;
                He[(size_t)mr * DFF + nc] = (bf16)fast_silu(acc[mi][ni][i] + bias);
            }
        }
}

// ---------------------------------------------------------------------------
// GEMM2: out[e,b,d] = sum_f H[e,b,f] * W2[e,d,f] + b2[e,d]. A already bf16.
// M=CAP (256), N=DMODEL (1024), K=DFF (4096).
// ---------------------------------------------------------------------------
__global__ __launch_bounds__(256)
void ffn_gemm2(const bf16* __restrict__ H,
               const float* __restrict__ W2,
               const float* __restrict__ B2,
               float* __restrict__ Out) {
    const int e    = blockIdx.z;
    const int nb   = blockIdx.x;          // N tile: DMODEL/BN
    const int mb   = blockIdx.y;          // M tile: CAP/BM
    const int tid  = threadIdx.x;
    const int lane = tid & 31;
    const int wave = tid >> 5;
    const int waveM = wave >> 1;
    const int waveN = wave & 1;
    const int khalf = lane >> 4;
    const int mrow  = lane & 15;

    __shared__ bf16 As[2][BM][LDA];
    __shared__ bf16 Bs[2][BN][LDA];

    const bf16*  He  = H  + (size_t)e * CAPN * DFF   + (size_t)mb * BM * DFF;
    const float* W2e = W2 + (size_t)e * DMODEL * DFF + (size_t)nb * BN * DFF;

    auto stage = [&](int buf, int k0) {
        // A (bf16, no convert): 128x32 bf16 = 512 uint4; 2 per thread.
#pragma unroll
        for (int it = 0; it < 2; ++it) {
            int idx = tid + it * 256;
            int r   = idx >> 2;                // 4 uint4 per 32-elem row
            int c   = idx & 3;
            uint4 v = ((const uint4*)(He + (size_t)r * DFF + k0))[c];
            *(uint4*)&As[buf][r][c * 8] = v;
        }
        // B: 64x32 fp32 -> bf16; 2 float4 each.
#pragma unroll
        for (int it = 0; it < 2; ++it) {
            int idx = tid + it * 256;
            int r   = idx >> 3;
            int c4  = idx & 7;
            float4 v = ((const float4*)(W2e + (size_t)r * DFF + k0))[c4];
            bf16* dst = &Bs[buf][r][c4 * 4];
            dst[0] = (bf16)v.x; dst[1] = (bf16)v.y;
            dst[2] = (bf16)v.z; dst[3] = (bf16)v.w;
        }
    };

    v8f acc[2][2] = {};

    stage(0, 0);
    __syncthreads();

    const int nsteps = DFF / BK;
    for (int ks = 0; ks < nsteps; ++ks) {
        const int buf = ks & 1;
        if (ks + 1 < nsteps) stage(buf ^ 1, (ks + 1) * BK);

        v16bf afrag[2], bfrag[2];
#pragma unroll
        for (int mi = 0; mi < 2; ++mi)
            afrag[mi] = load_a_frag(&As[buf][waveM * 32 + mi * 16 + mrow][0], khalf);
#pragma unroll
        for (int ni = 0; ni < 2; ++ni)
            bfrag[ni] = load_b_frag(&Bs[buf][waveN * 32 + ni * 16 + mrow][0], khalf);

#pragma unroll
        for (int mi = 0; mi < 2; ++mi)
#pragma unroll
            for (int ni = 0; ni < 2; ++ni)
                acc[mi][ni] = __builtin_amdgcn_wmma_f32_16x16x32_bf16(
                    false, afrag[mi], false, bfrag[ni],
                    (short)0, acc[mi][ni], false, false);
        __syncthreads();
    }

    const float* b2p = B2 + (size_t)e * DMODEL + (size_t)nb * BN;
    float* Oe = Out + (size_t)e * CAPN * DMODEL + (size_t)mb * BM * DMODEL
                    + (size_t)nb * BN;
    const int dm = (lane >> 4) * 8;
    const int dn = lane & 15;
#pragma unroll
    for (int mi = 0; mi < 2; ++mi)
#pragma unroll
        for (int ni = 0; ni < 2; ++ni) {
            int nc = waveN * 32 + ni * 16 + dn;
            float bias = b2p[nc];
#pragma unroll
            for (int i = 0; i < 8; ++i) {
                int mr = waveM * 32 + mi * 16 + dm + i;
                Oe[(size_t)mr * DMODEL + nc] = acc[mi][ni][i] + bias;
            }
        }
}

extern "C" void kernel_launch(void* const* d_in, const int* in_sizes, int n_in,
                              void* d_out, int out_size, void* d_ws, size_t ws_size,
                              hipStream_t stream) {
    const float* x  = (const float*)d_in[0];   // [E, CAP, D_MODEL]
    const float* w1 = (const float*)d_in[1];   // [E, D_FF, D_MODEL]
    const float* b1 = (const float*)d_in[2];   // [E, D_FF]
    const float* w2 = (const float*)d_in[3];   // [E, D_MODEL, D_FF]
    const float* b2 = (const float*)d_in[4];   // [E, D_MODEL]
    float* out = (float*)d_out;                // [E, CAP, D_MODEL]
    bf16* Hws  = (bf16*)d_ws;                  // [E, CAP, D_FF] bf16 = 32 MB

    dim3 blk(256);
    ffn_gemm1_silu<<<dim3(DFF / BN,    CAPN / BM, NEXPERT), blk, 0, stream>>>(x, w1, b1, Hws);
    ffn_gemm2     <<<dim3(DMODEL / BN, CAPN / BM, NEXPERT), blk, 0, stream>>>(Hws, w2, b2, out);
}
